// Association_44684839748139
// MI455X (gfx1250) — matrix operand order repaired and checked by reading, not compiled, not run on hardware
//
#include <hip/hip_runtime.h>

// ---------------------------------------------------------------------------
// Sparsemax attention for MI455X (gfx1250, wave32, WMMA + Tensor Data Mover).
//   scores = (Q K^T) / sqrt(E)   -> bf16-split WMMA (hi*hi + hi*lo + lo*hi)
//   A      = sparsemax(scores)   -> bisection on tau, register-resident rows
//   out    = A V                 -> bf16-split WMMA
// K and V tiles are staged to LDS with TDM (tensor_load_to_lds), double-
// buffered on TENSORcnt, so strided-global gathers become contiguous LDS reads.
// ---------------------------------------------------------------------------

typedef __attribute__((ext_vector_type(16))) __bf16        v16bf;
typedef __attribute__((ext_vector_type(8)))  float         v8f;
typedef __attribute__((ext_vector_type(4)))  unsigned int  v4u;
typedef __attribute__((ext_vector_type(8)))  int           v8i_;
typedef __attribute__((ext_vector_type(4)))  int           v4i_;

#define B_   8
#define L_   1024
#define S_   1024
#define H_   16
#define E_   64
#define D_   64

#define WAVES           2
#define ROWS_PER_WAVE   16
#define TILE_L          (WAVES * ROWS_PER_WAVE)   // 32 L-rows per workgroup
#define SSTRIDE         (S_ + 4)                  // pad rows across LDS banks

// LDS layout (floats)
#define SCORES_FLOATS   (TILE_L * SSTRIDE)                 // 32896
#define KSTAGE_FLOATS   (16 * E_)                          // 1024 per buffer
#define VSTAGE_FLOATS   (32 * D_)                          // 2048 per buffer
#define SMEM_FLOATS     (SCORES_FLOATS + WAVES * 2 * KSTAGE_FLOATS \
                                       + WAVES * 2 * VSTAGE_FLOATS)

static __device__ __forceinline__ v8f wmma_bf16(v16bf a, v16bf b, v8f c) {
    // 8 args: (neg_a, A, neg_b, B, c_mod, C, reuse_a, reuse_b)
    return __builtin_amdgcn_wmma_f32_16x16x32_bf16(
        false, a, false, b, (short)0, c, false, false);
}

static __device__ __forceinline__ void split2(float x, __bf16& h, __bf16& l) {
    h = (__bf16)x;
    l = (__bf16)(x - (float)h);
}

// A-matrix (16x32 bf16) per-lane K index for element t (contiguous order,
// matching the B fills):
//   lanes 0-15: K=0..15;  lanes16-31: K=16..31
static __device__ __forceinline__ int a_kidx(int t, int kh) {
    return kh * 16 + t;
}

// Issue a TDM 2D tile load: `rows` rows of `row_elems` f32, global row stride
// `row_stride_elems` f32, packed contiguously at LDS byte offset `lds_off`.
// D# packing per CDNA5 ISA ch.8 (group0: count/lds/global/type, group1: dims).
// This toolchain exposes the 6-arg builtin: (v4u, v8i, v4i, v4i, v8i, cpol).
static __device__ __forceinline__ void tdm_load_2d(unsigned lds_off,
                                                   const float* gptr,
                                                   int rows, int row_elems,
                                                   int row_stride_elems) {
    const unsigned long long ga = (unsigned long long)(size_t)gptr;
    v4u g0;
    g0[0] = 1u;                                        // count=1, user desc
    g0[1] = lds_off;                                   // lds_addr (bytes)
    g0[2] = (unsigned)(ga & 0xFFFFFFFFu);              // global_addr[31:0]
    g0[3] = (unsigned)((ga >> 32) & 0x1FFFFFFu)        // global_addr[56:32]
          | (2u << 30);                                // type=2 ("image")
    v8i_ g1;
    g1[0] = (int)(2u << 16);                           // data_size=2 -> 4B
    g1[1] = (int)(((unsigned)row_elems & 0xFFFFu) << 16);        // tdim0 lo
    g1[2] = (int)((((unsigned)row_elems >> 16) & 0xFFFFu)
          | (((unsigned)rows & 0xFFFFu) << 16));                 // tdim0 hi|tdim1 lo
    g1[3] = (int)((((unsigned)rows >> 16) & 0xFFFFu)
          | (((unsigned)row_elems & 0xFFFFu) << 16));            // tdim1 hi|tile_dim0
    g1[4] = (int)((unsigned)rows & 0xFFFFu);                     // tile_dim1 (tile_dim2=0)
    g1[5] = (int)(unsigned)row_stride_elems;                     // dim0_stride[31:0]
    g1[6] = 0;                                                   // stride hi, dim1_stride lo
    g1[7] = 0;
    const v4i_ z4 = {0, 0, 0, 0};                      // groups 2/3 unused (2D)
    const v8i_ z8 = {0, 0, 0, 0, 0, 0, 0, 0};
    __builtin_amdgcn_tensor_load_to_lds(g0, g1, z4, z4, z8, 0);
}

__global__ __launch_bounds__(WAVES * 32)
void sparsemax_attn_kernel(const float* __restrict__ q,
                           const float* __restrict__ kk,
                           const float* __restrict__ vv,
                           float* __restrict__ out)
{
    __shared__ float smem[SMEM_FLOATS];   // ~177 KB of the 320 KB WGP LDS

    const int lane = threadIdx.x & 31;
    const int wid  = threadIdx.x >> 5;
    const int b    = blockIdx.z;
    const int h    = blockIdx.y;
    const int l0   = blockIdx.x * TILE_L + wid * ROWS_PER_WAVE;

    const int ln = lane & 15;   // A: row M; B: column N; C/D: column N
    const int kh = lane >> 4;   // which K-half of the fragment this lane holds

    float* srow = smem + wid * ROWS_PER_WAVE * SSTRIDE;
    float* kst  = smem + SCORES_FLOATS + wid * 2 * KSTAGE_FLOATS;
    float* vst  = smem + SCORES_FLOATS + WAVES * 2 * KSTAGE_FLOATS
                       + wid * 2 * VSTAGE_FLOATS;
    const unsigned kst_off = (unsigned)(size_t)(void*)kst;   // LDS byte offset
    const unsigned vst_off = (unsigned)(size_t)(void*)vst;

    const float* kbase = kk + (((size_t)b * S_) * H_ + h) * E_;
    const float* vbase = vv + (((size_t)b * S_) * H_ + h) * D_;

    // ---------------- Pass 1: scores = (Q K^T) * scale -> LDS --------------
    // Q tile: 16 rows x 64 E, as two 16x32 bf16-split A fragments (reused).
    v16bf qa_hi[2], qa_lo[2];
    {
        const float* qrow = q + (((size_t)b * L_ + (l0 + ln)) * H_ + h) * E_;
        #pragma unroll
        for (int f = 0; f < 2; ++f) {
            #pragma unroll
            for (int t = 0; t < 16; ++t) {
                const float x = qrow[f * 32 + a_kidx(t, kh)];
                __bf16 hi, lo; split2(x, hi, lo);
                qa_hi[f][t] = hi; qa_lo[f][t] = lo;
            }
        }
    }

    const float scale = 0.125f;   // 1/sqrt(64)

    // Double-buffered TDM staging of K tiles: 16 rows x 64 f32, stride H*E.
    tdm_load_2d(kst_off, kbase, 16, E_, H_ * E_);
    for (int st = 0; st < S_ / 16; ++st) {
        if (st + 1 < S_ / 16) {
            tdm_load_2d(kst_off + ((st + 1) & 1) * KSTAGE_FLOATS * 4,
                        kbase + (size_t)(st + 1) * 16 * H_ * E_,
                        16, E_, H_ * E_);
            __builtin_amdgcn_s_wait_tensorcnt((short)1);
        } else {
            __builtin_amdgcn_s_wait_tensorcnt((short)0);
        }
        const float* kbuf = kst + (st & 1) * KSTAGE_FLOATS;

        v8f acc = {};
        #pragma unroll
        for (int f = 0; f < 2; ++f) {
            // B fragment: K^T tile 32(e) x 16(s); lane column s = st*16+ln,
            // contiguous e-run of 16 in the staged tile.
            v16bf kb_hi, kb_lo;
            #pragma unroll
            for (int t = 0; t < 16; ++t) {
                const float x = kbuf[ln * E_ + f * 32 + kh * 16 + t];
                __bf16 hi, lo; split2(x, hi, lo);
                kb_hi[t] = hi; kb_lo[t] = lo;
            }
            acc = wmma_bf16(qa_hi[f], kb_hi, acc);
            acc = wmma_bf16(qa_hi[f], kb_lo, acc);
            acc = wmma_bf16(qa_lo[f], kb_hi, acc);
        }
        // C/D layout: VGPR r -> row kh*8 + r, column ln.
        #pragma unroll
        for (int r = 0; r < 8; ++r)
            srow[(kh * 8 + r) * SSTRIDE + st * 16 + ln] = acc[r] * scale;
    }
    __syncthreads();

    // ---------------- Pass 2: sparsemax per row (bisection on tau) ---------
    for (int r = 0; r < ROWS_PER_WAVE; ++r) {
        float z[32];
        #pragma unroll
        for (int i = 0; i < 32; ++i) z[i] = srow[r * SSTRIDE + i * 32 + lane];

        float mx = z[0];
        #pragma unroll
        for (int i = 1; i < 32; ++i) mx = fmaxf(mx, z[i]);
        #pragma unroll
        for (int off = 16; off > 0; off >>= 1)
            mx = fmaxf(mx, __shfl_xor(mx, off, 32));

        // tau in [mx-1, mx); f(tau) = sum(max(z-tau,0)) - 1 is decreasing.
        float lo = mx - 1.0f, hi = mx;
        for (int it = 0; it < 24; ++it) {
            const float mid = 0.5f * (lo + hi);
            float s = 0.0f;
            #pragma unroll
            for (int i = 0; i < 32; ++i) s += fmaxf(z[i] - mid, 0.0f);
            #pragma unroll
            for (int off = 16; off > 0; off >>= 1)
                s += __shfl_xor(s, off, 32);
            const bool big = (s >= 1.0f);
            lo = big ? mid : lo;
            hi = big ? hi  : mid;
        }
        const float tau = 0.5f * (lo + hi);
        #pragma unroll
        for (int i = 0; i < 32; ++i)
            srow[r * SSTRIDE + i * 32 + lane] = fmaxf(z[i] - tau, 0.0f);
    }
    __syncthreads();

    // ---------------- Pass 3: out = A V (bf16-split WMMA) ------------------
    v8f acc2[4];
    #pragma unroll
    for (int dt = 0; dt < 4; ++dt) { v8f zr = {}; acc2[dt] = zr; }

    // Double-buffered TDM staging of V tiles: 32 rows x 64 f32, stride H*D.
    tdm_load_2d(vst_off, vbase, 32, D_, H_ * D_);
    for (int ks = 0; ks < S_ / 32; ++ks) {
        if (ks + 1 < S_ / 32) {
            tdm_load_2d(vst_off + ((ks + 1) & 1) * VSTAGE_FLOATS * 4,
                        vbase + (size_t)(ks + 1) * 32 * H_ * D_,
                        32, D_, H_ * D_);
            __builtin_amdgcn_s_wait_tensorcnt((short)1);
        } else {
            __builtin_amdgcn_s_wait_tensorcnt((short)0);
        }
        const float* vbuf = vst + (ks & 1) * VSTAGE_FLOATS;
        const int s0 = ks * 32;

        // A fragment: sparsemax weights, row ln, K-dim = s.
        v16bf aa_hi, aa_lo;
        #pragma unroll
        for (int t = 0; t < 16; ++t) {
            const float x = srow[ln * SSTRIDE + s0 + a_kidx(t, kh)];
            __bf16 hi, lo; split2(x, hi, lo);
            aa_hi[t] = hi; aa_lo[t] = lo;
        }
        #pragma unroll
        for (int dt = 0; dt < 4; ++dt) {
            // B fragment: staged V tile 32(s) x 16(d); lane column d=dt*16+ln.
            v16bf vb_hi, vb_lo;
            #pragma unroll
            for (int t = 0; t < 16; ++t) {
                const float x = vbuf[(kh * 16 + t) * D_ + dt * 16 + ln];
                __bf16 hi, lo; split2(x, hi, lo);
                vb_hi[t] = hi; vb_lo[t] = lo;
            }
            acc2[dt] = wmma_bf16(aa_hi, vb_hi, acc2[dt]);
            acc2[dt] = wmma_bf16(aa_hi, vb_lo, acc2[dt]);
            acc2[dt] = wmma_bf16(aa_lo, vb_hi, acc2[dt]);
        }
    }

    // Write out [B, L, H, D]; C/D layout: VGPR r -> row kh*8+r, col ln.
    #pragma unroll
    for (int dt = 0; dt < 4; ++dt) {
        #pragma unroll
        for (int r = 0; r < 8; ++r) {
            const int row = kh * 8 + r;
            out[(((size_t)b * L_ + (l0 + row)) * H_ + h) * D_ + dt * 16 + ln] =
                acc2[dt][r];
        }
    }
}

extern "C" void kernel_launch(void* const* d_in, const int* in_sizes, int n_in,
                              void* d_out, int out_size, void* d_ws, size_t ws_size,
                              hipStream_t stream) {
    (void)in_sizes; (void)n_in; (void)out_size; (void)d_ws; (void)ws_size;
    const float* q  = (const float*)d_in[0];
    const float* kk = (const float*)d_in[1];
    const float* vv = (const float*)d_in[2];
    float* out = (float*)d_out;

    dim3 grid(L_ / TILE_L, H_, B_);   // (32, 16, 8) workgroups
    dim3 block(WAVES * 32);           // 2 waves of 32
    sparsemax_attn_kernel<<<grid, block, 0, stream>>>(q, kk, vv, out);
}
